// AreaAttention_41721312314223
// MI455X (gfx1250) — matrix operand order repaired and checked by reading, not compiled, not run on hardware
//
#include <hip/hip_runtime.h>
#include <hip/hip_bf16.h>

typedef __bf16 bf16_t;
typedef __attribute__((ext_vector_type(16))) __bf16 v16bf;
typedef __attribute__((ext_vector_type(8)))  __bf16 v8bf;
typedef __attribute__((ext_vector_type(8)))  float  v8f;

#define EPS 1e-5f

__device__ __forceinline__ bf16_t f2bf(float f) {
    union { float f; unsigned u; } v; v.f = f;
    unsigned r = (v.u + 0x7FFFu + ((v.u >> 16) & 1u)) >> 16;
    unsigned short s = (unsigned short)r;
    return __builtin_bit_cast(bf16_t, s);
}

__device__ __forceinline__ float bn_silu(float y, float g, float b, float m, float v) {
    float s = g * rsqrtf(v + EPS);
    float t = (y - m) * s + b;
    return t / (1.0f + __expf(-t));
}

// ---- CDNA5 async global->LDS copy (per-lane 16B, tracked by ASYNCcnt) ----
// Generic shared pointers carry the LDS byte offset in addr[31:0] (flat aperture rule),
// so the low 32 bits are a valid VDST LDS address for the async op.
__device__ __forceinline__ void async_ld_b128(unsigned ldsoff, const void* g) {
    asm volatile("global_load_async_to_lds_b128 %0, %1, off"
                 :: "v"(ldsoff), "v"(g) : "memory");
}
__device__ __forceinline__ void wait_async0() {
    asm volatile("s_wait_asynccnt 0x0" ::: "memory");
}
__device__ __forceinline__ unsigned lds_off(const void* p) {
    return (unsigned)(size_t)p;
}

// ---- WMMA fragment loaders (bf16 16x16x32, wave32 layouts per CDNA5 ISA 7.12.2) ----
// A (16x32, MxK): lane m = lane&15; lanes<16 hold K {0..7,16..23}, lanes>=16 hold K {8..15,24..31}
__device__ __forceinline__ v16bf load_a_frag(const bf16_t* sA, int lane, int ld) {
    int m  = lane & 15;
    int kb = (lane >> 4) ? 8 : 0;
    const bf16_t* p = sA + m * ld;
    v16bf a;
#pragma unroll
    for (int i = 0; i < 8; ++i) a[i] = p[kb + i];
#pragma unroll
    for (int i = 0; i < 8; ++i) a[8 + i] = p[kb + 16 + i];
    return a;
}
// B (32x16, KxN) from LDS stored as Bt[n][k]: lane n = lane&15; lanes<16 hold K {0..7,16..23}, lanes>=16 hold K {8..15,24..31}
__device__ __forceinline__ v16bf load_b_frag(const bf16_t* sBt, int lane, int ld) {
    int n  = lane & 15;
    int kb = (lane >> 4) ? 8 : 0;
    const bf16_t* p = sBt + n * ld;
    v16bf b;
#pragma unroll
    for (int i = 0; i < 8; ++i) b[i] = p[kb + i];
#pragma unroll
    for (int i = 0; i < 8; ++i) b[8 + i] = p[kb + 16 + i];
    return b;
}

#define WMMA_BF16(A, B, C) \
    __builtin_amdgcn_wmma_f32_16x16x32_bf16(false, (A), false, (B), (short)0, (C), false, false)

// ------------------------------- small utility kernels -------------------------------
__global__ void k_f32_to_bf16(const float* __restrict__ s, bf16_t* __restrict__ d, int n) {
    int i = blockIdx.x * 256 + threadIdx.x;
    if (i < n) d[i] = f2bf(s[i]);
}

// pe_w [O=512][C=512][3][3] -> [tap(9)][O*C] bf16 (tap-major so each tap is a dense GEMM A)
__global__ void k_repack_pe(const float* __restrict__ s, bf16_t* __restrict__ d, int total) {
    int i = blockIdx.x * 256 + threadIdx.x;
    if (i >= total) return;
    int tap = i % 9;
    int oc  = i / 9;
    d[(size_t)tap * (512 * 512) + oc] = f2bf(s[i]);
}

__global__ void k_zero32(unsigned* __restrict__ p, int nwords) {
    int i = blockIdx.x * 256 + threadIdx.x;
    if (i < nwords) p[i] = 0u;
}

// attnOut [B][N][C] (f32) + pp [B][C][N] (f32) -> Zb [B][C][N] (bf16)
__global__ void k_add_tr(const float* __restrict__ ao, const float* __restrict__ pp,
                         bf16_t* __restrict__ z, int total) {
    int i = blockIdx.x * 256 + threadIdx.x;
    if (i >= total) return;
    int n = i & 4095;
    int c = (i >> 12) & 511;
    int b = i >> 21;
    float v = ao[(((size_t)b * 4096 + n) * 512) + c] + pp[i];
    z[i] = f2bf(v);
}

// ------------------------- fused QKV 1x1 conv + BN + SiLU + scatter -------------------------
// W [1536][512] bf16, X [8][512][4096] bf16.  Tile: M=128, N=64, K-step 32, 8 waves (4x2 of 32x32).
__global__ __launch_bounds__(256) void k_gemm_qkv(
    const bf16_t* __restrict__ W, const bf16_t* __restrict__ X,
    const float* __restrict__ qkg, const float* __restrict__ qkb,
    const float* __restrict__ qkm, const float* __restrict__ qkv,
    const float* __restrict__ vg, const float* __restrict__ vb,
    const float* __restrict__ vm, const float* __restrict__ vv,
    bf16_t* __restrict__ Qb, bf16_t* __restrict__ Kb, bf16_t* __restrict__ Vb,
    bf16_t* __restrict__ vmapP) {
    __shared__ bf16_t sA[128 * 32];
    __shared__ bf16_t sBt[64 * 32];
    const int t = threadIdx.x, lane = t & 31, w = t >> 5;
    const int wm = w & 3, wn = w >> 2;
    const int n0 = blockIdx.x * 64;
    const int m0 = blockIdx.y * 128;
    const int bidx = n0 >> 12;
    const int nb = n0 & 4095;

    v8f acc[2][2] = {};
    for (int k0 = 0; k0 < 512; k0 += 32) {
        { // stage A (row-major [m][k]) via async global->LDS copies
            int row = t >> 1, seg = (t & 1) << 4;
            const bf16_t* g = W + (size_t)(m0 + row) * 512 + k0 + seg;
            unsigned la = lds_off(sA + row * 32 + seg);
            async_ld_b128(la,      g);
            async_ld_b128(la + 16, g + 8);
        }
        { // stage Bt (transposed: [n][k]) -- needs VGPR round trip
            int k = t >> 3, nblk = (t & 7) << 3;
            const bf16_t* g = X + ((size_t)bidx * 512 + k0 + k) * 4096 + nb + nblk;
            v8bf vvv = *(const v8bf*)g;
#pragma unroll
            for (int i = 0; i < 8; ++i) sBt[(nblk + i) * 32 + k] = vvv[i];
            if (k0 + 32 < 512)
                __builtin_prefetch((const void*)(g + 32 * 4096), 0, 1);
        }
        wait_async0();
        __syncthreads();
        v16bf af[2], bfr[2];
#pragma unroll
        for (int mi = 0; mi < 2; ++mi) af[mi] = load_a_frag(sA + (wm * 32 + mi * 16) * 32, lane, 32);
#pragma unroll
        for (int ni = 0; ni < 2; ++ni) bfr[ni] = load_b_frag(sBt + (wn * 32 + ni * 16) * 32, lane, 32);
#pragma unroll
        for (int mi = 0; mi < 2; ++mi)
#pragma unroll
            for (int ni = 0; ni < 2; ++ni)
                acc[mi][ni] = WMMA_BF16(af[mi], bfr[ni], acc[mi][ni]);
        __syncthreads();
    }
    // epilogue: BN+SiLU, scatter to per-head Q/K/V layout [Ba][h][Na][64] and padded vmap
#pragma unroll
    for (int mi = 0; mi < 2; ++mi)
#pragma unroll
        for (int ni = 0; ni < 2; ++ni) {
            int nl = wn * 32 + ni * 16 + (lane & 15);
            int nn = nb + nl;
            int ba = (bidx << 2) + (nn >> 10);
            int na = nn & 1023;
#pragma unroll
            for (int v = 0; v < 8; ++v) {
                int o = m0 + wm * 32 + mi * 16 + ((lane >> 4) << 3) + v;
                float y = acc[mi][ni][v];
                float g, b, m, var;
                if (o < 1024) { g = qkg[o]; b = qkb[o]; m = qkm[o]; var = qkv[o]; }
                else { int oo = o - 1024; g = vg[oo]; b = vb[oo]; m = vm[oo]; var = vv[oo]; }
                float act = bn_silu(y, g, b, m, var);
                bf16_t hv = f2bf(act);
                if (o < 512) {
                    Qb[(((size_t)ba * 8 + (o >> 6)) * 1024 + na) * 64 + (o & 63)] = hv;
                } else if (o < 1024) {
                    int oo = o - 512;
                    Kb[(((size_t)ba * 8 + (oo >> 6)) * 1024 + na) * 64 + (oo & 63)] = hv;
                } else {
                    int oo = o - 1024;
                    Vb[(((size_t)ba * 8 + (oo >> 6)) * 1024 + na) * 64 + (oo & 63)] = hv;
                    int yy = nn >> 6, xx = nn & 63;
                    vmapP[(((size_t)bidx * 512 + oo) * 66 + yy + 1) * 66 + xx + 1] = hv;
                }
            }
        }
}

// ---------------- 3x3 conv as implicit GEMM over 9 taps + BN + SiLU -> pp (f32) ----------------
__global__ __launch_bounds__(256) void k_conv3x3(
    const bf16_t* __restrict__ W9,  // [9][512][512]
    const bf16_t* __restrict__ Vp,  // [8][512][66][66]
    const float* __restrict__ pg, const float* __restrict__ pb,
    const float* __restrict__ pm, const float* __restrict__ pv,
    float* __restrict__ pp) {
    __shared__ bf16_t sA[128 * 32];
    __shared__ bf16_t sBt[64 * 32];
    const int t = threadIdx.x, lane = t & 31, w = t >> 5;
    const int wm = w & 3, wn = w >> 2;
    const int y = blockIdx.x;          // image row, N-tile = one row of 64 pixels
    const int bidx = blockIdx.y;
    const int m0 = blockIdx.z * 128;

    v8f acc[2][2] = {};
    for (int tap = 0; tap < 9; ++tap) {
        const int ky = tap / 3, kx = tap % 3;
        const bf16_t* Wt = W9 + (size_t)tap * 512 * 512;
        const bf16_t* src0 = Vp + (((size_t)bidx * 512) * 66 + (y + ky)) * 66 + kx;
        for (int k0 = 0; k0 < 512; k0 += 32) {
            { // async stage of weight tile
                int row = t >> 1, seg = (t & 1) << 4;
                const bf16_t* g = Wt + (size_t)(m0 + row) * 512 + k0 + seg;
                unsigned la = lds_off(sA + row * 32 + seg);
                async_ld_b128(la,      g);
                async_ld_b128(la + 16, g + 8);
            }
            { // transposed stage of activation tile (2B-aligned source; scalar loads)
                int k = t >> 3, nblk = (t & 7) << 3;
                const bf16_t* g = src0 + (size_t)(k0 + k) * (66 * 66) + nblk;
#pragma unroll
                for (int i = 0; i < 8; ++i) sBt[(nblk + i) * 32 + k] = g[i];
            }
            wait_async0();
            __syncthreads();
            v16bf af[2], bfr[2];
#pragma unroll
            for (int mi = 0; mi < 2; ++mi) af[mi] = load_a_frag(sA + (wm * 32 + mi * 16) * 32, lane, 32);
#pragma unroll
            for (int ni = 0; ni < 2; ++ni) bfr[ni] = load_b_frag(sBt + (wn * 32 + ni * 16) * 32, lane, 32);
#pragma unroll
            for (int mi = 0; mi < 2; ++mi)
#pragma unroll
                for (int ni = 0; ni < 2; ++ni)
                    acc[mi][ni] = WMMA_BF16(af[mi], bfr[ni], acc[mi][ni]);
            __syncthreads();
        }
    }
#pragma unroll
    for (int mi = 0; mi < 2; ++mi)
#pragma unroll
        for (int ni = 0; ni < 2; ++ni) {
            int x = wn * 32 + ni * 16 + (lane & 15);
#pragma unroll
            for (int v = 0; v < 8; ++v) {
                int o = m0 + wm * 32 + mi * 16 + ((lane >> 4) << 3) + v;
                float act = bn_silu(acc[mi][ni][v], pg[o], pb[o], pm[o], pv[o]);
                pp[((size_t)bidx * 512 + o) * 4096 + y * 64 + x] = act;
            }
        }
}

// ------------------------- flash attention: 32 area-batches x 8 heads, hd=64 -------------------------
__global__ __launch_bounds__(256) void k_attn(
    const bf16_t* __restrict__ Qb, const bf16_t* __restrict__ Kb,
    const bf16_t* __restrict__ Vb, float* __restrict__ out) {
    __shared__ bf16_t sQ[128 * 64];
    __shared__ bf16_t sK[64 * 64];
    __shared__ bf16_t sVt[64 * 64];     // transposed: [d][key]
    __shared__ bf16_t sP[8 * 16 * 64];  // per-wave P staging
    const int t = threadIdx.x, lane = t & 31, w = t >> 5;
    const int qt = blockIdx.x;  // 0..7 (128 queries each)
    const int h  = blockIdx.y;  // 0..7
    const int ba = blockIdx.z;  // 0..31
    const size_t headOff = ((size_t)ba * 8 + h) * 1024 * 64;
    const bf16_t* Qh = Qb + headOff + (size_t)qt * 128 * 64;
    const bf16_t* Kh = Kb + headOff;
    const bf16_t* Vh = Vb + headOff;

    {   // stage Q tile 128x64 via async copies (no transpose needed)
        int row = t >> 1, seg = (t & 1) << 5;
        const bf16_t* g = Qh + row * 64 + seg;
        unsigned la = lds_off(sQ + row * 64 + seg);
        async_ld_b128(la,      g);
        async_ld_b128(la + 16, g + 8);
        async_ld_b128(la + 32, g + 16);
        async_ld_b128(la + 48, g + 24);
        wait_async0();
    }
    __syncthreads();

    float m_i[8], l_i[8];
#pragma unroll
    for (int v = 0; v < 8; ++v) { m_i[v] = -1e30f; l_i[v] = 0.0f; }
    v8f accO[4] = {};
    const float scale = 0.125f;  // 64^-0.5

    for (int kc = 0; kc < 1024; kc += 64) {
        {   // stage K chunk [key][d] via async (natural layout == Bt layout for Q*K^T)
            int row = t >> 2, seg = (t & 3) << 4;
            const bf16_t* g = Kh + (size_t)(kc + row) * 64 + seg;
            unsigned la = lds_off(sK + row * 64 + seg);
            async_ld_b128(la,      g);
            async_ld_b128(la + 16, g + 8);
        }
        {   // stage V chunk transposed -> sVt[d][key] (transpose: VGPR round trip)
            int key = t >> 2, seg = (t & 3) << 4;
            const bf16_t* g = Vh + (size_t)(kc + key) * 64 + seg;
            v8bf a0 = *(const v8bf*)g, a1 = *(const v8bf*)(g + 8);
#pragma unroll
            for (int i = 0; i < 8; ++i) sVt[(seg + i) * 64 + key] = a0[i];
#pragma unroll
            for (int i = 0; i < 8; ++i) sVt[(seg + 8 + i) * 64 + key] = a1[i];
        }
        wait_async0();
        __syncthreads();

        // S = Q K^T (16 q-rows per wave x 64 keys)
        v8f accS[4] = {};
#pragma unroll
        for (int ks = 0; ks < 64; ks += 32) {
            v16bf a = load_a_frag(sQ + (w * 16) * 64 + ks, lane, 64);
#pragma unroll
            for (int nt = 0; nt < 4; ++nt) {
                v16bf b = load_b_frag(sK + (nt * 16) * 64 + ks, lane, 64);
                accS[nt] = WMMA_BF16(a, b, accS[nt]);
            }
        }
        // online softmax over the 64-key chunk (row = acc element + lane-half, cols across 16 lanes)
#pragma unroll
        for (int v = 0; v < 8; ++v) {
#pragma unroll
            for (int nt = 0; nt < 4; ++nt) accS[nt][v] *= scale;
            float c0 = fmaxf(fmaxf(accS[0][v], accS[1][v]), fmaxf(accS[2][v], accS[3][v]));
#pragma unroll
            for (int off = 1; off < 16; off <<= 1) c0 = fmaxf(c0, __shfl_xor(c0, off, 32));
            float mn = fmaxf(m_i[v], c0);
            float al = __expf(m_i[v] - mn);
            float rs = 0.0f;
#pragma unroll
            for (int nt = 0; nt < 4; ++nt) {
                float e = __expf(accS[nt][v] - mn);
                accS[nt][v] = e;
                rs += e;
            }
#pragma unroll
            for (int off = 1; off < 16; off <<= 1) rs += __shfl_xor(rs, off, 32);
            l_i[v] = l_i[v] * al + rs;
            m_i[v] = mn;
#pragma unroll
            for (int dt = 0; dt < 4; ++dt) accO[dt][v] *= al;
        }
        // transpose P (C-layout -> A-layout) through per-wave LDS
        bf16_t* sPw = sP + w * 16 * 64;
        int prow = (lane >> 4) << 3;
        int pcol = lane & 15;
#pragma unroll
        for (int nt = 0; nt < 4; ++nt)
#pragma unroll
            for (int v = 0; v < 8; ++v)
                sPw[(prow + v) * 64 + pcol + nt * 16] = f2bf(accS[nt][v]);
        asm volatile("s_wait_dscnt 0x0" ::: "memory");
        // O += P * V
#pragma unroll
        for (int ks = 0; ks < 64; ks += 32) {
            v16bf a = load_a_frag(sPw + ks, lane, 64);
#pragma unroll
            for (int dt = 0; dt < 4; ++dt) {
                v16bf b = load_b_frag(sVt + (dt * 16) * 64 + ks, lane, 64);
                accO[dt] = WMMA_BF16(a, b, accO[dt]);
            }
        }
        __syncthreads();
    }
    // normalize and store to attnOut [Ba][Na][C] f32
    const size_t outBase = ((size_t)ba * 1024 + qt * 128 + w * 16) * 512 + h * 64;
    int rbase = (lane >> 4) << 3;
#pragma unroll
    for (int dt = 0; dt < 4; ++dt) {
        int d = (lane & 15) + dt * 16;
#pragma unroll
        for (int v = 0; v < 8; ++v)
            out[outBase + (size_t)(rbase + v) * 512 + d] = accO[dt][v] / l_i[v];
    }
}

// ------------------- final 1x1 conv (pr) + BN + SiLU -> d_out [B][C][H][W] f32 -------------------
__global__ __launch_bounds__(256) void k_gemm_out(
    const bf16_t* __restrict__ W, const bf16_t* __restrict__ X,
    const float* __restrict__ pg, const float* __restrict__ pb,
    const float* __restrict__ pm, const float* __restrict__ pv,
    float* __restrict__ Y) {
    __shared__ bf16_t sA[128 * 32];
    __shared__ bf16_t sBt[64 * 32];
    const int t = threadIdx.x, lane = t & 31, w = t >> 5;
    const int wm = w & 3, wn = w >> 2;
    const int n0 = blockIdx.x * 64;
    const int m0 = blockIdx.y * 128;
    const int bidx = n0 >> 12;
    const int nb = n0 & 4095;

    v8f acc[2][2] = {};
    for (int k0 = 0; k0 < 512; k0 += 32) {
        {
            int row = t >> 1, seg = (t & 1) << 4;
            const bf16_t* g = W + (size_t)(m0 + row) * 512 + k0 + seg;
            unsigned la = lds_off(sA + row * 32 + seg);
            async_ld_b128(la,      g);
            async_ld_b128(la + 16, g + 8);
        }
        {
            int k = t >> 3, nblk = (t & 7) << 3;
            const bf16_t* g = X + ((size_t)bidx * 512 + k0 + k) * 4096 + nb + nblk;
            v8bf vvv = *(const v8bf*)g;
#pragma unroll
            for (int i = 0; i < 8; ++i) sBt[(nblk + i) * 32 + k] = vvv[i];
            if (k0 + 32 < 512)
                __builtin_prefetch((const void*)(g + 32 * 4096), 0, 1);
        }
        wait_async0();
        __syncthreads();
        v16bf af[2], bfr[2];
#pragma unroll
        for (int mi = 0; mi < 2; ++mi) af[mi] = load_a_frag(sA + (wm * 32 + mi * 16) * 32, lane, 32);
#pragma unroll
        for (int ni = 0; ni < 2; ++ni) bfr[ni] = load_b_frag(sBt + (wn * 32 + ni * 16) * 32, lane, 32);
#pragma unroll
        for (int mi = 0; mi < 2; ++mi)
#pragma unroll
            for (int ni = 0; ni < 2; ++ni)
                acc[mi][ni] = WMMA_BF16(af[mi], bfr[ni], acc[mi][ni]);
        __syncthreads();
    }
#pragma unroll
    for (int mi = 0; mi < 2; ++mi)
#pragma unroll
        for (int ni = 0; ni < 2; ++ni) {
            int nn = nb + wn * 32 + ni * 16 + (lane & 15);
#pragma unroll
            for (int v = 0; v < 8; ++v) {
                int o = m0 + wm * 32 + mi * 16 + ((lane >> 4) << 3) + v;
                float act = bn_silu(acc[mi][ni][v], pg[o], pb[o], pm[o], pv[o]);
                Y[((size_t)bidx * 512 + o) * 4096 + nn] = act;
            }
        }
}

// ------------------------------------- host launcher -------------------------------------
extern "C" void kernel_launch(void* const* d_in, const int* in_sizes, int n_in,
                              void* d_out, int out_size, void* d_ws, size_t ws_size,
                              hipStream_t stream) {
    (void)in_sizes; (void)n_in; (void)out_size; (void)ws_size;
    const float* x     = (const float*)d_in[0];
    const float* qk_w  = (const float*)d_in[1];
    const float* qk_g  = (const float*)d_in[2];
    const float* qk_b  = (const float*)d_in[3];
    const float* qk_rm = (const float*)d_in[4];
    const float* qk_rv = (const float*)d_in[5];
    const float* v_w   = (const float*)d_in[6];
    const float* v_g   = (const float*)d_in[7];
    const float* v_b   = (const float*)d_in[8];
    const float* v_rm  = (const float*)d_in[9];
    const float* v_rv  = (const float*)d_in[10];
    const float* pe_w  = (const float*)d_in[11];
    const float* pe_g  = (const float*)d_in[12];
    const float* pe_b  = (const float*)d_in[13];
    const float* pe_rm = (const float*)d_in[14];
    const float* pe_rv = (const float*)d_in[15];
    const float* pr_w  = (const float*)d_in[16];
    const float* pr_g  = (const float*)d_in[17];
    const float* pr_b  = (const float*)d_in[18];
    const float* pr_rm = (const float*)d_in[19];
    const float* pr_rv = (const float*)d_in[20];

    size_t off = 0;
    char* base = (char*)d_ws;
    auto alloc = [&](size_t bytes) -> void* {
        void* p = base + off;
        off += (bytes + 255) & ~(size_t)255;
        return p;
    };
    const size_t NTOT = 8ull * 512 * 4096;            // 16,777,216
    bf16_t* Xb    = (bf16_t*)alloc(NTOT * 2);         // x in bf16 [B][C][N]
    bf16_t* Wqkv  = (bf16_t*)alloc(1536ull * 512 * 2);
    bf16_t* Wpe9  = (bf16_t*)alloc(9ull * 512 * 512 * 2);
    bf16_t* Wpr   = (bf16_t*)alloc(512ull * 512 * 2);
    bf16_t* Qb    = (bf16_t*)alloc(NTOT * 2);         // [Ba][h][Na][64]
    bf16_t* Kb    = (bf16_t*)alloc(NTOT * 2);
    bf16_t* Vb    = (bf16_t*)alloc(NTOT * 2);
    bf16_t* vmapP = (bf16_t*)alloc(8ull * 512 * 66 * 66 * 2);
    float*  attnO = (float*)alloc(NTOT * 4);          // [B][N][C]
    float*  pp    = (float*)alloc(NTOT * 4);          // [B][C][N]
    bf16_t* Zb    = (bf16_t*)alloc(NTOT * 2);         // [B][C][N]

    // precision conversion / repack
    k_f32_to_bf16<<<(int)((NTOT + 255) / 256), 256, 0, stream>>>(x, Xb, (int)NTOT);
    k_f32_to_bf16<<<(524288 + 255) / 256, 256, 0, stream>>>(qk_w, Wqkv, 524288);
    k_f32_to_bf16<<<(262144 + 255) / 256, 256, 0, stream>>>(v_w, Wqkv + 524288, 262144);
    k_repack_pe<<<(2359296 + 255) / 256, 256, 0, stream>>>(pe_w, Wpe9, 2359296);
    k_f32_to_bf16<<<(262144 + 255) / 256, 256, 0, stream>>>(pr_w, Wpr, 262144);
    {
        int nwords = (8 * 512 * 66 * 66) / 2;
        k_zero32<<<(nwords + 255) / 256, 256, 0, stream>>>((unsigned*)vmapP, nwords);
    }
    // fused qk+v GEMM -> Q/K/V + padded vmap
    k_gemm_qkv<<<dim3(512, 12), 256, 0, stream>>>(Wqkv, Xb,
        qk_g, qk_b, qk_rm, qk_rv, v_g, v_b, v_rm, v_rv, Qb, Kb, Vb, vmapP);
    // 3x3 conv on vmap -> pp
    k_conv3x3<<<dim3(64, 8, 4), 256, 0, stream>>>(Wpe9, vmapP,
        pe_g, pe_b, pe_rm, pe_rv, pp);
    // flash attention
    k_attn<<<dim3(8, 8, 32), 256, 0, stream>>>(Qb, Kb, Vb, attnO);
    // (out + pp) -> bf16 [B][C][N]
    k_add_tr<<<(int)((NTOT + 255) / 256), 256, 0, stream>>>(attnO, pp, Zb, (int)NTOT);
    // final 1x1 conv + BN + SiLU -> d_out
    k_gemm_out<<<dim3(512, 4), 256, 0, stream>>>(Wpr, Zb,
        pr_g, pr_b, pr_rm, pr_rv, (float*)d_out);
}
